// Attention_63977832841510
// MI455X (gfx1250) — matrix operand order repaired and checked by reading, not compiled, not run on hardware
//
#include <hip/hip_runtime.h>
#include <hip/hip_bf16.h>
#include <math.h>

// Problem constants (from reference): B=8, C=256, H=W=40
#define NBATCH 8
#define NC     256
#define NHW    40
#define NL     1600      // H*W
#define NHEAD  4
#define DH     64        // C / NHEAD
#define DKQ    32        // DH / 2
#define C2     512       // 2*C
#define EPSBN  0.001f
#define QSCALE 0.17677669529663689f   // dk^-0.5 = 1/sqrt(32)

typedef __bf16 v16bf __attribute__((ext_vector_type(16)));
typedef float  v8f   __attribute__((ext_vector_type(8)));

union ABfrag { v16bf v; unsigned u[8]; };
union CDfrag { v8f   v; float    f[8]; };

// round-to-nearest-even f32 -> bf16 bits
__device__ __forceinline__ unsigned short f2bf(float x) {
    union { float f; unsigned u; } c; c.f = x;
    unsigned r = c.u + 0x7FFFu + ((c.u >> 16) & 1u);
    return (unsigned short)(r >> 16);
}

// A-fragment K index for vgpr j, half-wave h (16-bit A, 16x32): ISA 7.12.2
__device__ __forceinline__ int kA(int j, int half) {
    return ((j >> 2) << 4) + ((j & 3) << 1) + (half << 3);
}
// B-fragment K index for vgpr j, half-wave h (16-bit B, 32x16)
__device__ __forceinline__ int kB(int j, int half) {
    return ((j >> 2) << 4) + ((j & 3) << 1) + (half << 3);
}

// ---------------------------------------------------------------------------
// Prep: fold BN into weights, convert to bf16; make folded biases
// ---------------------------------------------------------------------------
__global__ void prep_weights(const float* __restrict__ qkv_w, const float* __restrict__ qkv_g,
                             const float* __restrict__ qkv_b, const float* __restrict__ qkv_m,
                             const float* __restrict__ qkv_v,
                             const float* __restrict__ cb2_w, const float* __restrict__ cb2_g,
                             const float* __restrict__ cb2_b, const float* __restrict__ cb2_m,
                             const float* __restrict__ cb2_v,
                             unsigned short* __restrict__ Wq, float* __restrict__ biasq,
                             unsigned short* __restrict__ W2, float* __restrict__ bias2) {
    int i = blockIdx.x * 256 + threadIdx.x;
    const int nWq = C2 * NC, nW2 = NC * NC;
    if (i < nWq) {
        int o = i / NC;
        float inv = qkv_g[o] * rsqrtf(qkv_v[o] + EPSBN);
        Wq[i] = f2bf(qkv_w[i] * inv);
    } else if (i < nWq + nW2) {
        int j = i - nWq;
        int o = j / NC;
        float inv = cb2_g[o] * rsqrtf(cb2_v[o] + EPSBN);
        W2[j] = f2bf(cb2_w[j] * inv);
    } else if (i < nWq + nW2 + C2) {
        int o = i - nWq - nW2;
        float inv = qkv_g[o] * rsqrtf(qkv_v[o] + EPSBN);
        biasq[o] = qkv_b[o] - qkv_m[o] * inv;
    } else if (i < nWq + nW2 + C2 + NC) {
        int o = i - nWq - nW2 - C2;
        float inv = cb2_g[o] * rsqrtf(cb2_v[o] + EPSBN);
        bias2[o] = cb2_b[o] - cb2_m[o] * inv;
    }
}

// Prep: x [b,c,l] f32 -> xt [b,l,c] bf16 (contiguous-c for B fragments)
__global__ void prep_x(const float* __restrict__ x, unsigned short* __restrict__ xt) {
    int i = blockIdx.x * 256 + threadIdx.x;
    if (i >= NBATCH * NC * NL) return;
    int l = i % NL;
    int c = (i / NL) % NC;
    int b = i / (NL * NC);
    xt[((size_t)b * NL + l) * NC + c] = f2bf(x[i]);
}

// ---------------------------------------------------------------------------
// QKV projection GEMM: (512x256)x(256x1600) per batch, BN folded.
// Each wave owns a 16(o) x 64(l) strip: 4 N-tiles share one A fragment per
// K-step (reuse-A hint on the identical back-to-back WMMAs).
// q: [b,h,l,32] bf16 (pre-scaled)   k: [b,h,l,32] bf16
// v: [b,h,64,l] bf16  +  v: [b,c,l] f32 (for depthwise conv)
// ---------------------------------------------------------------------------
__global__ void __launch_bounds__(128) qkv_gemm(
        const unsigned short* __restrict__ Wq, const float* __restrict__ biasq,
        const unsigned short* __restrict__ xt,
        unsigned short* __restrict__ qb, unsigned short* __restrict__ kb,
        unsigned short* __restrict__ vb, float* __restrict__ vf) {
    int wave = (blockIdx.x * 128 + threadIdx.x) >> 5;
    int lane = threadIdx.x & 31;
    int b   = wave / (32 * 25);
    int rem = wave % (32 * 25);
    int ot = rem / 25, lt4 = rem % 25;
    int l0 = lt4 * 64;
    int m = lane & 15, half = lane >> 4;

    v8f z = {};
    CDfrag acc[4];
    acc[0].v = z; acc[1].v = z; acc[2].v = z; acc[3].v = z;

    const unsigned short* arow = Wq + (size_t)(ot * 16 + m) * NC;
    const unsigned short* brow = xt + ((size_t)b * NL + l0 + m) * NC;

    for (int kc = 0; kc < NC; kc += 32) {
        ABfrag a;
#pragma unroll
        for (int j = 0; j < 8; ++j)
            a.u[j] = *(const unsigned*)(arow + kc + kA(j, half));
        ABfrag bm0, bm1, bm2, bm3;
#pragma unroll
        for (int j = 0; j < 8; ++j) {
            int kk = kc + kB(j, half);
            bm0.u[j] = *(const unsigned*)(brow + (size_t)(0 * 16) * NC + kk);
            bm1.u[j] = *(const unsigned*)(brow + (size_t)(1 * 16) * NC + kk);
            bm2.u[j] = *(const unsigned*)(brow + (size_t)(2 * 16) * NC + kk);
            bm3.u[j] = *(const unsigned*)(brow + (size_t)(3 * 16) * NC + kk);
        }
        acc[0].v = __builtin_amdgcn_wmma_f32_16x16x32_bf16(false, a.v, false, bm0.v, (short)0, acc[0].v, false, false);
        acc[1].v = __builtin_amdgcn_wmma_f32_16x16x32_bf16(false, a.v, false, bm1.v, (short)0, acc[1].v, true,  false);
        acc[2].v = __builtin_amdgcn_wmma_f32_16x16x32_bf16(false, a.v, false, bm2.v, (short)0, acc[2].v, true,  false);
        acc[3].v = __builtin_amdgcn_wmma_f32_16x16x32_bf16(false, a.v, false, bm3.v, (short)0, acc[3].v, true,  false);
    }

#pragma unroll
    for (int t = 0; t < 4; ++t) {
        int l = l0 + t * 16 + m;
#pragma unroll
        for (int r = 0; r < 8; ++r) {
            int o = ot * 16 + r + half * 8;
            float val = acc[t].f[r] + biasq[o];
            int head = o >> 7;      // 128 channels per head in qkv order
            int oin  = o & 127;
            if (oin < DKQ) {
                qb[((size_t)(b * NHEAD + head) * NL + l) * DKQ + oin] = f2bf(val * QSCALE);
            } else if (oin < 2 * DKQ) {
                kb[((size_t)(b * NHEAD + head) * NL + l) * DKQ + (oin - DKQ)] = f2bf(val);
            } else {
                int d = oin - 2 * DKQ;
                vb[((size_t)(b * NHEAD + head) * DH + d) * NL + l] = f2bf(val);
                vf[((size_t)b * NC + head * DH + d) * NL + l] = val;
            }
        }
    }
}

// ---------------------------------------------------------------------------
// Flash attention: per wave one (b, head, 16-row q tile); K chunks of 32.
// S via 2 WMMAs, online softmax via half-wave shuffles, P through LDS to
// A-layout, O += P*V via 4 WMMAs (dh=64). Writes y f32 [b,c,l].
// ---------------------------------------------------------------------------
__global__ void __launch_bounds__(128) attn_kernel(
        const unsigned short* __restrict__ qb, const unsigned short* __restrict__ kb,
        const unsigned short* __restrict__ vb, float* __restrict__ y) {
    __shared__ __align__(16) unsigned short P[4][16][32];
    int wid  = threadIdx.x >> 5;
    int wave = (blockIdx.x * 128 + threadIdx.x) >> 5;
    int lane = threadIdx.x & 31;
    int qt = wave % 100;
    int h  = (wave / 100) % NHEAD;
    int b  = wave / (100 * NHEAD);
    int m = lane & 15, half = lane >> 4;

    // Q tile stays resident as A-fragment for the whole K loop
    ABfrag qa;
    const unsigned short* qrow = qb + ((size_t)(b * NHEAD + h) * NL + qt * 16 + m) * DKQ;
#pragma unroll
    for (int j = 0; j < 8; ++j) qa.u[j] = *(const unsigned*)(qrow + kA(j, half));

    float mi[8], li[8];
    CDfrag o0, o1, o2, o3;
#pragma unroll
    for (int r = 0; r < 8; ++r) {
        mi[r] = -3.0e38f; li[r] = 0.f;
        o0.f[r] = o1.f[r] = o2.f[r] = o3.f[r] = 0.f;
    }

    const unsigned short* kbase = kb + (size_t)(b * NHEAD + h) * NL * DKQ;
    const unsigned short* vbase = vb + (size_t)(b * NHEAD + h) * DH * NL;
    v8f z = {};

    for (int kc = 0; kc < NL; kc += 32) {
        // S = q . k  (two 16-wide kpos tiles, K = dk = 32 in one WMMA each)
        ABfrag b0, b1;
#pragma unroll
        for (int j = 0; j < 8; ++j) {
            int kk = kB(j, half);
            b0.u[j] = *(const unsigned*)(kbase + (size_t)(kc + m) * DKQ + kk);
            b1.u[j] = *(const unsigned*)(kbase + (size_t)(kc + 16 + m) * DKQ + kk);
        }
        CDfrag s0, s1;
        s0.v = __builtin_amdgcn_wmma_f32_16x16x32_bf16(false, qa.v, false, b0.v, (short)0, z, false, false);
        s1.v = __builtin_amdgcn_wmma_f32_16x16x32_bf16(false, qa.v, false, b1.v, (short)0, z, true, false);

        // Online softmax: rows live in the 16-lane half; xor 1/2/4/8 reduces a row
#pragma unroll
        for (int r = 0; r < 8; ++r) {
            float t = fmaxf(s0.f[r], s1.f[r]);
            t = fmaxf(t, __shfl_xor(t, 1));
            t = fmaxf(t, __shfl_xor(t, 2));
            t = fmaxf(t, __shfl_xor(t, 4));
            t = fmaxf(t, __shfl_xor(t, 8));
            float mn = fmaxf(mi[r], t);
            float alpha = __expf(mi[r] - mn);
            mi[r] = mn;
            float p0 = __expf(s0.f[r] - mn);
            float p1 = __expf(s1.f[r] - mn);
            float rs = p0 + p1;
            rs += __shfl_xor(rs, 1);
            rs += __shfl_xor(rs, 2);
            rs += __shfl_xor(rs, 4);
            rs += __shfl_xor(rs, 8);
            li[r] = li[r] * alpha + rs;
            o0.f[r] *= alpha; o1.f[r] *= alpha; o2.f[r] *= alpha; o3.f[r] *= alpha;
            int row = r + half * 8;
            P[wid][row][m]      = f2bf(p0);
            P[wid][row][16 + m] = f2bf(p1);
        }

        // C/D layout -> A layout via LDS (per-wave slice; DS ops in-order per wave)
        ABfrag pa;
#pragma unroll
        for (int j = 0; j < 8; ++j)
            pa.u[j] = *(const unsigned*)(&P[wid][m][kA(j, half)]);

        // O += P * V   (dh = 64 -> four N=16 tiles, shared A => reuse hints)
        ABfrag v0, v1, v2, v3;
#pragma unroll
        for (int j = 0; j < 8; ++j) {
            int kk = kc + kB(j, half);
            v0.u[j] = *(const unsigned*)(vbase + (size_t)(0 * 16 + m) * NL + kk);
            v1.u[j] = *(const unsigned*)(vbase + (size_t)(1 * 16 + m) * NL + kk);
            v2.u[j] = *(const unsigned*)(vbase + (size_t)(2 * 16 + m) * NL + kk);
            v3.u[j] = *(const unsigned*)(vbase + (size_t)(3 * 16 + m) * NL + kk);
        }
        o0.v = __builtin_amdgcn_wmma_f32_16x16x32_bf16(false, pa.v, false, v0.v, (short)0, o0.v, false, false);
        o1.v = __builtin_amdgcn_wmma_f32_16x16x32_bf16(false, pa.v, false, v1.v, (short)0, o1.v, true,  false);
        o2.v = __builtin_amdgcn_wmma_f32_16x16x32_bf16(false, pa.v, false, v2.v, (short)0, o2.v, true,  false);
        o3.v = __builtin_amdgcn_wmma_f32_16x16x32_bf16(false, pa.v, false, v3.v, (short)0, o3.v, true,  false);
    }

    // Epilogue: O /= rowsum, write y [b, h*64+d, qpos]
    float* ybase = y + ((size_t)b * NC + h * DH) * NL;
#pragma unroll
    for (int r = 0; r < 8; ++r) {
        float invl = 1.0f / li[r];
        int qpos = qt * 16 + r + half * 8;
        ybase[(size_t)(0 * 16 + m) * NL + qpos] = o0.f[r] * invl;
        ybase[(size_t)(1 * 16 + m) * NL + qpos] = o1.f[r] * invl;
        ybase[(size_t)(2 * 16 + m) * NL + qpos] = o2.f[r] * invl;
        ybase[(size_t)(3 * 16 + m) * NL + qpos] = o3.f[r] * invl;
    }
}

// ---------------------------------------------------------------------------
// Depthwise 3x3 conv on v + BN + residual add; writes yt [b,l,c] bf16
// ---------------------------------------------------------------------------
__global__ void conv_bn_add(const float* __restrict__ vf, const float* __restrict__ cb1_w,
                            const float* __restrict__ g, const float* __restrict__ bb,
                            const float* __restrict__ mm, const float* __restrict__ vv,
                            const float* __restrict__ y, unsigned short* __restrict__ yt) {
    int i = blockIdx.x * 256 + threadIdx.x;
    if (i >= NBATCH * NC * NL) return;
    int l = i % NL;
    int c = (i / NL) % NC;
    int b = i / (NL * NC);
    int px = l % NHW, py = l / NHW;
    const float* vch = vf + ((size_t)b * NC + c) * NL;
    const float* wk  = cb1_w + c * 9;
    float s = 0.f;
#pragma unroll
    for (int dy = -1; dy <= 1; ++dy)
#pragma unroll
        for (int dx = -1; dx <= 1; ++dx) {
            int yy = py + dy, xx = px + dx;
            if (yy >= 0 && yy < NHW && xx >= 0 && xx < NHW)
                s += vch[yy * NHW + xx] * wk[(dy + 1) * 3 + (dx + 1)];
        }
    float inv = g[c] * rsqrtf(vv[c] + EPSBN);
    float val = y[i] + s * inv + (bb[c] - mm[c] * inv);
    yt[((size_t)b * NL + l) * NC + c] = f2bf(val);
}

// ---------------------------------------------------------------------------
// Output projection GEMM (256x256) + folded BN -> d_out f32 [b,c,l]
// 16(o) x 64(l) strip per wave, shared A fragment + reuse hints.
// ---------------------------------------------------------------------------
__global__ void __launch_bounds__(128) proj_gemm(
        const unsigned short* __restrict__ W2, const float* __restrict__ bias2,
        const unsigned short* __restrict__ yt, float* __restrict__ out) {
    int wave = (blockIdx.x * 128 + threadIdx.x) >> 5;
    int lane = threadIdx.x & 31;
    int b   = wave / (16 * 25);
    int rem = wave % (16 * 25);
    int ot = rem / 25, lt4 = rem % 25;
    int l0 = lt4 * 64;
    int m = lane & 15, half = lane >> 4;

    v8f z = {};
    CDfrag acc[4];
    acc[0].v = z; acc[1].v = z; acc[2].v = z; acc[3].v = z;

    const unsigned short* arow = W2 + (size_t)(ot * 16 + m) * NC;
    const unsigned short* brow = yt + ((size_t)b * NL + l0 + m) * NC;

    for (int kc = 0; kc < NC; kc += 32) {
        ABfrag a;
#pragma unroll
        for (int j = 0; j < 8; ++j)
            a.u[j] = *(const unsigned*)(arow + kc + kA(j, half));
        ABfrag bm0, bm1, bm2, bm3;
#pragma unroll
        for (int j = 0; j < 8; ++j) {
            int kk = kc + kB(j, half);
            bm0.u[j] = *(const unsigned*)(brow + (size_t)(0 * 16) * NC + kk);
            bm1.u[j] = *(const unsigned*)(brow + (size_t)(1 * 16) * NC + kk);
            bm2.u[j] = *(const unsigned*)(brow + (size_t)(2 * 16) * NC + kk);
            bm3.u[j] = *(const unsigned*)(brow + (size_t)(3 * 16) * NC + kk);
        }
        acc[0].v = __builtin_amdgcn_wmma_f32_16x16x32_bf16(false, a.v, false, bm0.v, (short)0, acc[0].v, false, false);
        acc[1].v = __builtin_amdgcn_wmma_f32_16x16x32_bf16(false, a.v, false, bm1.v, (short)0, acc[1].v, true,  false);
        acc[2].v = __builtin_amdgcn_wmma_f32_16x16x32_bf16(false, a.v, false, bm2.v, (short)0, acc[2].v, true,  false);
        acc[3].v = __builtin_amdgcn_wmma_f32_16x16x32_bf16(false, a.v, false, bm3.v, (short)0, acc[3].v, true,  false);
    }

#pragma unroll
    for (int t = 0; t < 4; ++t) {
        int l = l0 + t * 16 + m;
#pragma unroll
        for (int r = 0; r < 8; ++r) {
            int o = ot * 16 + r + half * 8;
            out[((size_t)b * NC + o) * NL + l] = acc[t].f[r] + bias2[o];
        }
    }
}

// ---------------------------------------------------------------------------
extern "C" void kernel_launch(void* const* d_in, const int* in_sizes, int n_in,
                              void* d_out, int out_size, void* d_ws, size_t ws_size,
                              hipStream_t stream) {
    (void)in_sizes; (void)n_in; (void)out_size; (void)ws_size;

    const float* x      = (const float*)d_in[0];
    const float* qkv_w  = (const float*)d_in[1];
    const float* qkv_g  = (const float*)d_in[2];
    const float* qkv_b  = (const float*)d_in[3];
    const float* qkv_m  = (const float*)d_in[4];
    const float* qkv_v  = (const float*)d_in[5];
    const float* cb1_w  = (const float*)d_in[6];
    const float* cb1_g  = (const float*)d_in[7];
    const float* cb1_b  = (const float*)d_in[8];
    const float* cb1_m  = (const float*)d_in[9];
    const float* cb1_v  = (const float*)d_in[10];
    const float* cb2_w  = (const float*)d_in[11];
    const float* cb2_g  = (const float*)d_in[12];
    const float* cb2_b  = (const float*)d_in[13];
    const float* cb2_m  = (const float*)d_in[14];
    const float* cb2_v  = (const float*)d_in[15];
    float* out = (float*)d_out;

    // Workspace carve (256B aligned)
    char* ws = (char*)d_ws;
    size_t off = 0;
    auto carve = [&](size_t bytes) {
        void* p = ws + off;
        off = (off + bytes + 255) & ~(size_t)255;
        return p;
    };
    unsigned short* Wq    = (unsigned short*)carve((size_t)C2 * NC * 2);      // 256 KB
    float*          biasq = (float*)carve((size_t)C2 * 4);
    unsigned short* W2    = (unsigned short*)carve((size_t)NC * NC * 2);      // 128 KB
    float*          bias2 = (float*)carve((size_t)NC * 4);
    unsigned short* xt    = (unsigned short*)carve((size_t)NBATCH * NL * NC * 2);        // 6.5 MB
    unsigned short* qb    = (unsigned short*)carve((size_t)NBATCH * NHEAD * NL * DKQ * 2); // 3.3 MB
    unsigned short* kb    = (unsigned short*)carve((size_t)NBATCH * NHEAD * NL * DKQ * 2); // 3.3 MB
    unsigned short* vb    = (unsigned short*)carve((size_t)NBATCH * NHEAD * DH * NL * 2);  // 6.5 MB
    float*          vf    = (float*)carve((size_t)NBATCH * NC * NL * 4);                   // 13.1 MB
    float*          yf    = (float*)carve((size_t)NBATCH * NC * NL * 4);                   // 13.1 MB
    unsigned short* yt    = (unsigned short*)carve((size_t)NBATCH * NL * NC * 2);          // 6.5 MB

    // 1. BN-folded weight conversion + biases
    {
        int n = C2 * NC + NC * NC + C2 + NC;
        prep_weights<<<(n + 255) / 256, 256, 0, stream>>>(
            qkv_w, qkv_g, qkv_b, qkv_m, qkv_v,
            cb2_w, cb2_g, cb2_b, cb2_m, cb2_v,
            Wq, biasq, W2, bias2);
    }
    // 2. x transpose -> bf16 [b,l,c]
    prep_x<<<(NBATCH * NC * NL + 255) / 256, 256, 0, stream>>>(x, xt);

    // 3. QKV projection GEMM (WMMA): 6400 strip-waves (16x64 out each)
    qkv_gemm<<<1600, 128, 0, stream>>>(Wq, biasq, xt, qb, kb, vb, vf);

    // 4. Flash attention (WMMA): 3200 tile-waves
    attn_kernel<<<800, 128, 0, stream>>>(qb, kb, vb, yf);

    // 5. Depthwise conv + BN + residual, write transposed bf16
    conv_bn_add<<<(NBATCH * NC * NL + 255) / 256, 256, 0, stream>>>(
        vf, cb1_w, cb1_g, cb1_b, cb1_m, cb1_v, yf, yt);

    // 6. Output projection GEMM (WMMA) + BN -> d_out
    proj_gemm<<<800, 128, 0, stream>>>(W2, bias2, yt, out);
}
